// PoseEncoder_2508260901536
// MI455X (gfx1250) — matrix operand order repaired and checked by reading, not compiled, not run on hardware
//
#include <hip/hip_runtime.h>

typedef __attribute__((ext_vector_type(16))) _Float16 v16h;
typedef __attribute__((ext_vector_type(8)))  float    v8f;

#define BN_EPS 1e-5f
#define GEMM_WAVES 4

// ---------------------------------------------------------------------------
// Weight prep: W (O x C) f32 -> packed WMMA B fragments, zero padded to Kp.
// Block (ntile*kt + ktile) holds 512 halves: [lane][e] with
//   k = ktile*32 + (lane>>4)*16 + e ,  n = ntile*16 + (lane&15)
// so each wave reads its fragment as 32 contiguous bytes per lane.
// ---------------------------------------------------------------------------
__global__ void k_pack_weight(const float* __restrict__ W, _Float16* __restrict__ Bp,
                              int O, int C, int Kp) {
  int t = blockIdx.x * blockDim.x + threadIdx.x;
  int kt = Kp >> 5;
  int total = (O >> 4) * kt * 512;
  if (t >= total) return;
  int e    = t & 15;
  int lane = (t >> 4) & 31;
  int blk  = t >> 9;
  int ntile = blk / kt, ktile = blk - ntile * kt;
  int n = (ntile << 4) + (lane & 15);
  int k = (ktile << 5) + ((lane >> 4) << 4) + e;
  Bp[t] = (k < C) ? (_Float16)W[n * C + k] : (_Float16)0.f;
}

// ---------------------------------------------------------------------------
// WMMA GEMM: Y(MxN,f32) = A(MxK,f16 row-major) * Bp(packed fragments).
// Each wave computes a 16x32 tile (two WMMAs per A fragment). Fused per-channel
// sum/sumsq for BatchNorm. K mult of 32, N mult of 32, M mult of 16.
// ---------------------------------------------------------------------------
__global__ void k_gemm_wmma(const _Float16* __restrict__ A, const _Float16* __restrict__ Bp,
                            float* __restrict__ Y, float* __restrict__ stats,
                            int M, int K, int N) {
  const int wave = threadIdx.x >> 5;
  const int lane = threadIdx.x & 31;
  const int tm = (blockIdx.x * GEMM_WAVES + wave) * 16;
  const int tn = blockIdx.y * 32;
  if (tm >= M) return;                      // wave-uniform guard: EXEC all-ones

  const int col = lane & 15;
  const int hi  = lane >> 4;
  const int kt  = K >> 5;
  const _Float16* Ar = A + (size_t)(tm + col) * K;
  const _Float16* b0 = Bp + (((size_t)(tn >> 4) * kt) << 9) + (lane << 4);
  const _Float16* b1 = b0 + ((size_t)kt << 9);
  v8f acc0 = {}, acc1 = {};

  for (int k0 = 0; k0 < K; k0 += 32) {
    v16h a, b;
    const _Float16* ap = Ar + k0 + (hi << 4);
#pragma unroll
    for (int e = 0; e < 8; ++e) a[e] = ap[e];
#pragma unroll
    for (int e = 0; e < 8; ++e) a[8 + e] = ap[8 + e];
#pragma unroll
    for (int e = 0; e < 16; ++e) b[e] = b0[e];
    acc0 = __builtin_amdgcn_wmma_f32_16x16x32_f16(false, a, false, b,
                                                  (short)0, acc0, false, false);
#pragma unroll
    for (int e = 0; e < 16; ++e) b[e] = b1[e];
    acc1 = __builtin_amdgcn_wmma_f32_16x16x32_f16(false, a, false, b,
                                                  (short)0, acc1, false, false);
    b0 += 512; b1 += 512;
  }

  // D layout: acc[r] -> row tm + 8*hi + r, col (lane&15) within tile
  float* yp = Y + (size_t)(tm + (hi << 3)) * N + tn + col;
  float s10 = 0.f, s20 = 0.f, s11 = 0.f, s21 = 0.f;
#pragma unroll
  for (int r = 0; r < 8; ++r) {
    float v0 = acc0[r], v1 = acc1[r];
    yp[(size_t)r * N]      = v0;
    yp[(size_t)r * N + 16] = v1;
    s10 += v0; s20 += v0 * v0;
    s11 += v1; s21 += v1 * v1;
  }
  if (stats != nullptr) {
    atomicAdd(&stats[tn + col],          s10);
    atomicAdd(&stats[N + tn + col],      s20);
    atomicAdd(&stats[tn + 16 + col],     s11);
    atomicAdd(&stats[N + tn + 16 + col], s21);
  }
}

// ---------------------------------------------------------------------------
// BatchNorm fold + apply (linear bias cancels exactly under BN, never applied)
// ---------------------------------------------------------------------------
__global__ void k_bn_prep(const float* __restrict__ stats, const float* __restrict__ g,
                          const float* __restrict__ beta, float invM,
                          float* __restrict__ scsh, int N) {
  int i = blockIdx.x * blockDim.x + threadIdx.x;
  if (i >= N) return;
  float m = stats[i] * invM;
  float v = stats[N + i] * invM - m * m;
  float sc = g[i] * rsqrtf(v + BN_EPS);
  scsh[i] = sc;
  scsh[N + i] = beta[i] - m * sc;
}

__global__ void k_bn_apply(float* __restrict__ Y, const float* __restrict__ scsh,
                           _Float16* __restrict__ Ah, int M, int N) {
  size_t t = (size_t)blockIdx.x * blockDim.x + threadIdx.x;
  if (t >= (size_t)M * N) return;
  int c = (int)(t % N);
  float v = fmaxf(Y[t] * scsh[c] + scsh[N + c], 0.f);
  Y[t] = v;
  Ah[t] = (_Float16)v;
}

__global__ void k_maxpool(const float* __restrict__ Y, float* __restrict__ out,
                          int rows_out, int group, int N) {
  int t = blockIdx.x * blockDim.x + threadIdx.x;
  if (t >= rows_out * N) return;
  int r = t / N, c = t - r * N;
  const float* p = Y + (size_t)r * group * N + c;
  float m = p[0];
  for (int j = 1; j < group; ++j) m = fmaxf(m, p[(size_t)j * N]);
  out[t] = m;
}

// ---------------------------------------------------------------------------
// Farthest point sampling: one block (256 threads) per batch, distances in
// registers, LDS tree argmax, S sequential rounds.
// ---------------------------------------------------------------------------
template <int PPT>
__global__ void k_fps(const float* __restrict__ xyz, float* __restrict__ new_xyz,
                      int N, int S) {
  const int b = blockIdx.x, tid = threadIdx.x;
  const float* X = xyz + (size_t)b * N * 3;
  float dist[PPT];
#pragma unroll
  for (int i = 0; i < PPT; ++i) dist[i] = 1e10f;
  __shared__ float sv[256];
  __shared__ int   si[256];
  __shared__ float cent[3];
  int farthest = 0;
  for (int it = 0; it < S; ++it) {
    if (tid == 0) {
      float cx = X[farthest * 3 + 0], cy = X[farthest * 3 + 1], cz = X[farthest * 3 + 2];
      cent[0] = cx; cent[1] = cy; cent[2] = cz;
      float* nx = new_xyz + ((size_t)b * S + it) * 3;
      nx[0] = cx; nx[1] = cy; nx[2] = cz;
    }
    __syncthreads();
    float cx = cent[0], cy = cent[1], cz = cent[2];
    float best = -1.f; int bi = 0;
#pragma unroll
    for (int i = 0; i < PPT; ++i) {
      int p = tid * PPT + i;
      float dx = X[p * 3 + 0] - cx, dy = X[p * 3 + 1] - cy, dz = X[p * 3 + 2] - cz;
      float d = dx * dx + dy * dy + dz * dz;
      dist[i] = fminf(dist[i], d);
      if (dist[i] > best) { best = dist[i]; bi = p; }
    }
    sv[tid] = best; si[tid] = bi;
    __syncthreads();
    for (int st = 128; st > 0; st >>= 1) {
      if (tid < st && sv[tid + st] > sv[tid]) { sv[tid] = sv[tid + st]; si[tid] = si[tid + st]; }
      __syncthreads();
    }
    farthest = si[0];
  }
}

// ---------------------------------------------------------------------------
// Ball query: one wave32 per query; ordered first-ns-within-radius via ballot
// + prefix-popcount compaction. Pads with first hit.
// ---------------------------------------------------------------------------
__global__ void k_ball_query(const float* __restrict__ xyz, const float* __restrict__ new_xyz,
                             int* __restrict__ idx, int B, int N, int S, float r2, int ns) {
  const int lane = threadIdx.x & 31;
  const int wid = (blockIdx.x * blockDim.x + threadIdx.x) >> 5;
  if (wid >= B * S) return;
  const int b = wid / S, s = wid - b * S;
  const float* X = xyz + (size_t)b * N * 3;
  const float* q = new_xyz + ((size_t)b * S + s) * 3;
  float qx = q[0], qy = q[1], qz = q[2];
  int* out = idx + (size_t)wid * ns;
  int cnt = 0, first = 0;
  bool haveFirst = false;
  for (int base = 0; base < N && cnt < ns; base += 32) {
    int p = base + lane;
    float dx = X[p * 3 + 0] - qx, dy = X[p * 3 + 1] - qy, dz = X[p * 3 + 2] - qz;
    float d = dx * dx + dy * dy + dz * dz;
    bool in = (d <= r2);
    unsigned mask = (unsigned)__ballot(in);
    if (!haveFirst && mask) {
      first = __shfl(p, __ffs((int)mask) - 1);
      haveFirst = true;
    }
    int rank = __popc((int)(mask & ((1u << lane) - 1u)));
    int pos = cnt + rank;
    if (in && pos < ns) out[pos] = p;
    cnt += __popc((int)mask);
  }
  int c0 = cnt < ns ? cnt : ns;
  for (int j = c0 + lane; j < ns; j += 32) out[j] = first;
}

__global__ void k_group_gather(const float* __restrict__ pts, const int* __restrict__ idx,
                               _Float16* __restrict__ Ah, int Kp,
                               int B, int Np, int C, int S, int ns) {
  int row = blockIdx.x * blockDim.x + threadIdx.x;
  int rows = B * S * ns;
  if (row >= rows) return;
  int b = row / (S * ns);
  int j = row - b * (S * ns);
  int pi = idx[(size_t)b * S * ns + j];
  const float* src = pts + ((size_t)b * Np + pi) * C;
  _Float16* ar = Ah + (size_t)row * Kp;
  for (int c = 0; c < C; ++c) ar[c] = (_Float16)src[c];
  for (int c = C; c < Kp; ++c) ar[c] = (_Float16)0.f;
}

__global__ void k_cast_f32_f16(const float* __restrict__ X, _Float16* __restrict__ A, int n) {
  int t = blockIdx.x * blockDim.x + threadIdx.x;
  if (t < n) A[t] = (_Float16)X[t];
}

__global__ void k_concat_bcast(const float* __restrict__ pts1, int C1,
                               const float* __restrict__ pts2, int C2,
                               _Float16* __restrict__ Ah, int Kp, int B, int N) {
  int t = blockIdx.x * blockDim.x + threadIdx.x;
  if (t >= B * N) return;
  int b = t / N;
  _Float16* ar = Ah + (size_t)t * Kp;
  const float* p1 = pts1 + (size_t)t * C1;
  const float* p2 = pts2 + (size_t)b * C2;
  for (int c = 0; c < C1; ++c) ar[c] = (_Float16)p1[c];
  for (int c = 0; c < C2; ++c) ar[C1 + c] = (_Float16)p2[c];
  for (int c = C1 + C2; c < Kp; ++c) ar[c] = (_Float16)0.f;
}

__global__ void k_interp_concat(const float* __restrict__ xyz1, const float* __restrict__ xyz2,
                                const float* __restrict__ pts1, int C1,
                                const float* __restrict__ pts2, int C2,
                                _Float16* __restrict__ Ah, int Kp,
                                int B, int N, int S2) {
  int t = blockIdx.x * blockDim.x + threadIdx.x;
  if (t >= B * N) return;
  int b = t / N;
  const float* X2 = xyz2 + (size_t)b * S2 * 3;
  const float* q = xyz1 + (size_t)t * 3;
  float px = q[0], py = q[1], pz = q[2];
  float d0 = 3e38f, d1 = 3e38f, d2 = 3e38f;
  int i0 = 0, i1 = 0, i2 = 0;
  for (int s = 0; s < S2; ++s) {
    float dx = X2[s * 3 + 0] - px, dy = X2[s * 3 + 1] - py, dz = X2[s * 3 + 2] - pz;
    float d = dx * dx + dy * dy + dz * dz;
    if (d < d0)      { d2 = d1; i2 = i1; d1 = d0; i1 = i0; d0 = d; i0 = s; }
    else if (d < d1) { d2 = d1; i2 = i1; d1 = d; i1 = s; }
    else if (d < d2) { d2 = d; i2 = s; }
  }
  float w0 = 1.f / (d0 + 1e-8f), w1 = 1.f / (d1 + 1e-8f), w2 = 1.f / (d2 + 1e-8f);
  float ws = w0 + w1 + w2; w0 /= ws; w1 /= ws; w2 /= ws;
  _Float16* ar = Ah + (size_t)t * Kp;
  const float* p1 = pts1 + (size_t)t * C1;
  for (int c = 0; c < C1; ++c) ar[c] = (_Float16)p1[c];
  const float* a0 = pts2 + ((size_t)b * S2 + i0) * C2;
  const float* a1 = pts2 + ((size_t)b * S2 + i1) * C2;
  const float* a2 = pts2 + ((size_t)b * S2 + i2) * C2;
  for (int c = 0; c < C2; ++c)
    ar[C1 + c] = (_Float16)(w0 * a0[c] + w1 * a1[c] + w2 * a2[c]);
  for (int c = C1 + C2; c < Kp; ++c) ar[c] = (_Float16)0.f;
}

__global__ void k_out_transpose(const float* __restrict__ Y, const float* __restrict__ bias,
                                float* __restrict__ out, int B, int N, int C) {
  int t = blockIdx.x * blockDim.x + threadIdx.x;
  if (t >= B * N * C) return;
  int b = t / (N * C);
  int rem = t - b * (N * C);
  int o = rem / N, n = rem - o * N;
  out[t] = Y[((size_t)b * N + n) * C + o] + bias[o];
}

static const int kO[18] = {32,32,64, 64,64,128, 128,128,256, 256,256, 256,256, 256,128,128,128, 128};
static const int kC[18] = {3, 32,32, 64,64,64,  128,128,128, 384,256, 320,256, 259,256,128,128, 128};

extern "C" void kernel_launch(void* const* d_in, const int* in_sizes, int n_in,
                              void* d_out, int out_size, void* d_ws, size_t ws_size,
                              hipStream_t stream) {
  (void)in_sizes; (void)n_in; (void)out_size; (void)ws_size;
  const float* xyz = (const float*)d_in[0];   // d_in[1] unused: l0_points = xyz

  struct LW { const float* w; const float* g; const float* be; _Float16* Bh; int O, C, Kp; };
  LW L[18];
  int p = 2;
  for (int i = 0; i < 18; ++i) {
    L[i].w  = (const float*)d_in[p + 0];
    L[i].g  = (const float*)d_in[p + 2];
    L[i].be = (const float*)d_in[p + 3];
    p += 4;
    L[i].O = kO[i]; L[i].C = kC[i]; L[i].Kp = ((kC[i] + 31) / 32) * 32;
  }
  const float* w2 = (const float*)d_in[p + 0];
  const float* b2 = (const float*)d_in[p + 1];

  char* ws = (char*)d_ws; size_t off = 0;
  auto alloc = [&](size_t bytes) -> void* {
    void* r = ws + off; off = (off + bytes + 255) & ~(size_t)255; return r;
  };

  float* stats = (float*)alloc(512 * 4);
  float* scsh  = (float*)alloc(512 * 4);
  float* fps1  = (float*)alloc((size_t)8 * 512 * 3 * 4);
  float* fps2  = (float*)alloc((size_t)8 * 128 * 3 * 4);
  int*   idx1  = (int*)  alloc((size_t)8 * 512 * 32 * 4);
  int*   idx2  = (int*)  alloc((size_t)8 * 128 * 32 * 4);
  float* l1p   = (float*)alloc((size_t)8 * 512 * 64 * 4);
  float* l2p   = (float*)alloc((size_t)8 * 128 * 128 * 4);
  float* l3p   = (float*)alloc((size_t)8 * 256 * 4);
  float* l2p2  = (float*)alloc((size_t)8 * 128 * 256 * 4);
  float* l1p2  = (float*)alloc((size_t)8 * 512 * 256 * 4);
  float*    bufY  = (float*)   alloc((size_t)65536 * 256 * 4);
  _Float16* bufA  = (_Float16*)alloc((size_t)65536 * 288 * 2);
  _Float16* bufA2 = (_Float16*)alloc((size_t)65536 * 256 * 2);

  for (int i = 0; i < 18; ++i) {
    L[i].Bh = (_Float16*)alloc((size_t)L[i].Kp * L[i].O * 2);
    int tot = L[i].Kp * L[i].O;
    k_pack_weight<<<(tot + 255) / 256, 256, 0, stream>>>(L[i].w, L[i].Bh, L[i].O, L[i].C, L[i].Kp);
  }
  _Float16* Bh2 = (_Float16*)alloc((size_t)128 * 128 * 2);
  k_pack_weight<<<(128 * 128 + 255) / 256, 256, 0, stream>>>(w2, Bh2, 128, 128, 128);

  auto run_layer = [&](const LW& lw, _Float16* Ain, _Float16* Aout, int M) {
    int N = lw.O, K = lw.Kp;
    hipMemsetAsync(stats, 0, 2 * N * sizeof(float), stream);
    dim3 grid((M + 16 * GEMM_WAVES - 1) / (16 * GEMM_WAVES), N / 32);
    k_gemm_wmma<<<grid, 32 * GEMM_WAVES, 0, stream>>>(Ain, lw.Bh, bufY, stats, M, K, N);
    k_bn_prep<<<(N + 63) / 64, 64, 0, stream>>>(stats, lw.g, lw.be, 1.0f / (float)M, scsh, N);
    size_t tot = (size_t)M * N;
    k_bn_apply<<<(unsigned)((tot + 255) / 256), 256, 0, stream>>>(bufY, scsh, Aout, M, N);
  };
  auto run_stack = [&](int first, int n, int M) -> _Float16* {
    _Float16* cur = bufA; _Float16* alt = bufA2;
    for (int i = 0; i < n; ++i) {
      run_layer(L[first + i], cur, alt, M);
      _Float16* t = cur; cur = alt; alt = t;
    }
    return cur;
  };

  // SA1: 8192 -> 512 centers, C=3 -> [32,32,64], maxpool over 32
  k_fps<32><<<8, 256, 0, stream>>>(xyz, fps1, 8192, 512);
  k_ball_query<<<512, 256, 0, stream>>>(xyz, fps1, idx1, 8, 8192, 512, 0.04f, 32);
  k_group_gather<<<(131072 + 255) / 256, 256, 0, stream>>>(xyz, idx1, bufA, 32, 8, 8192, 3, 512, 32);
  run_stack(0, 3, 131072);
  k_maxpool<<<(4096 * 64 + 255) / 256, 256, 0, stream>>>(bufY, l1p, 4096, 32, 64);

  // SA2: 512 -> 128 centers, C=64 -> [64,64,128]
  k_fps<2><<<8, 256, 0, stream>>>(fps1, fps2, 512, 128);
  k_ball_query<<<128, 256, 0, stream>>>(fps1, fps2, idx2, 8, 512, 128, 0.16f, 32);
  k_group_gather<<<(32768 + 255) / 256, 256, 0, stream>>>(l1p, idx2, bufA, 64, 8, 512, 64, 128, 32);
  run_stack(3, 3, 32768);
  k_maxpool<<<(1024 * 128 + 255) / 256, 256, 0, stream>>>(bufY, l2p, 1024, 32, 128);

  // SA3: global MLP [128,128,256], maxpool over 128
  k_cast_f32_f16<<<(1024 * 128 + 255) / 256, 256, 0, stream>>>(l2p, bufA, 1024 * 128);
  run_stack(6, 3, 1024);
  k_maxpool<<<(8 * 256 + 255) / 256, 256, 0, stream>>>(bufY, l3p, 8, 128, 256);

  // FP3 (S==1 broadcast): concat[l2(128), l3(256)] -> [256,256]
  k_concat_bcast<<<(1024 + 255) / 256, 256, 0, stream>>>(l2p, 128, l3p, 256, bufA, 384, 8, 128);
  run_stack(9, 2, 1024);
  hipMemcpyAsync(l2p2, bufY, (size_t)1024 * 256 * 4, hipMemcpyDeviceToDevice, stream);

  // FP2: 3-NN(512 <- 128), concat[l1(64), interp(256)] -> [256,256]
  k_interp_concat<<<(4096 + 255) / 256, 256, 0, stream>>>(fps1, fps2, l1p, 64, l2p2, 256,
                                                          bufA, 320, 8, 512, 128);
  run_stack(11, 2, 4096);
  hipMemcpyAsync(l1p2, bufY, (size_t)4096 * 256 * 4, hipMemcpyDeviceToDevice, stream);

  // FP1 + head1: 3-NN(8192 <- 512), concat[xyz(3), interp(256)] pad->288,
  // MLP [256,128,128,128] + head1 [128]
  k_interp_concat<<<(65536 + 255) / 256, 256, 0, stream>>>(xyz, fps1, xyz, 3, l1p2, 256,
                                                           bufA, 288, 8, 8192, 512);
  _Float16* hf = run_stack(13, 5, 65536);

  // head2: plain linear 128->128 + bias, transpose to (B,128,N)
  dim3 g2(65536 / (16 * GEMM_WAVES), 128 / 32);
  k_gemm_wmma<<<g2, 32 * GEMM_WAVES, 0, stream>>>(hf, Bh2, bufY, nullptr, 65536, 128, 128);
  k_out_transpose<<<(8 * 8192 * 128 + 255) / 256, 256, 0, stream>>>(bufY, b2, (float*)d_out,
                                                                    8, 8192, 128);
}
